// CombinedLoss_2276332667354
// MI455X (gfx1250) — matrix operand order repaired and checked by reading, not compiled, not run on hardware
//
#include <hip/hip_runtime.h>

#define MARGIN 1.0f
#define EPS_PD 1e-6f

typedef __attribute__((ext_vector_type(16))) __bf16 v16bf;
typedef __attribute__((ext_vector_type(8)))  float  v8f;

union BFrag { v16bf v; uint4 q[2]; };

__device__ __forceinline__ unsigned short f32_to_bf16_rne(float f) {
  unsigned int u = __float_as_uint(f);
  unsigned int r = 0x7FFFu + ((u >> 16) & 1u);
  return (unsigned short)((u + r) >> 16);
}
__device__ __forceinline__ float bf16_to_f32(unsigned short h) {
  return __uint_as_float(((unsigned int)h) << 16);
}

// ---------------- kernel 0: zero accumulators ----------------
__global__ void zero_acc_kernel(double* acc) {
  acc[0] = 0.0;  // contrastive sum
  acc[1] = 0.0;  // supervised sum
}

// ---------------- kernel 1: split f32 -> bf16 hi/lo ----------------
__global__ void split_bf16_kernel(const float* __restrict__ x,
                                  unsigned short* __restrict__ hi,
                                  unsigned short* __restrict__ lo,
                                  int n) {
  int i = blockIdx.x * blockDim.x + threadIdx.x;
  if (i >= n) return;
  float v = x[i];
  unsigned short h = f32_to_bf16_rne(v);
  hi[i] = h;
  lo[i] = f32_to_bf16_rne(v - bf16_to_f32(h));
}

// ---------------- kernel 2: per-row sum and sum-of-squares ----------------
__global__ void row_stats_kernel(const float* __restrict__ L,
                                 float* __restrict__ sq,
                                 float* __restrict__ sv, int D) {
  __shared__ float r0[256];
  __shared__ float r1[256];
  int row = blockIdx.x;
  const float* p = L + (size_t)row * D;
  float a = 0.f, b = 0.f;
  for (int c = threadIdx.x; c < D; c += blockDim.x) {
    float x = p[c];
    a += x;
    b += x * x;
  }
  r0[threadIdx.x] = a;
  r1[threadIdx.x] = b;
  __syncthreads();
  for (int off = 128; off > 0; off >>= 1) {
    if ((int)threadIdx.x < off) {
      r0[threadIdx.x] += r0[threadIdx.x + off];
      r1[threadIdx.x] += r1[threadIdx.x + off];
    }
    __syncthreads();
  }
  if (threadIdx.x == 0) {
    sv[row] = r0[0];
    sq[row] = r1[0];
  }
}

// ---------------- kernel 3: WMMA gram tiles + pair loss ----------------
// One wave computes a 16(M) x 64(N) strip: 4 accumulators, bf16x3 emulated
// f32 GEMM over K=D via v_wmma_f32_16x16x32_bf16.
__global__ void __launch_bounds__(256)
contrastive_kernel(const unsigned short* __restrict__ Hi,
                   const unsigned short* __restrict__ Lo,
                   const float* __restrict__ sq,
                   const float* __restrict__ sv,
                   const int* __restrict__ labels,
                   double* __restrict__ acc,
                   int N, int D, float dEps2) {
  const int wave  = (int)((blockIdx.x * blockDim.x + threadIdx.x) >> 5);
  const int lane  = (int)(threadIdx.x & 31);
  const int gcols = N >> 6;              // column groups of 64
  const int rows  = N >> 4;              // row tiles of 16
  const int rt = wave / gcols;
  const int cg = wave - rt * gcols;
  if (rt >= rows) return;                // wave-uniform exit
  const int i0 = rt << 4;
  const int j0 = cg << 6;
  if (i0 >= j0 + 63) return;             // no (i<j) in this block; uniform

  const int m  = lane & 15;              // A row / B col within tile
  const int hh = lane >> 4;              // lane half

  const unsigned short* rAh = Hi + (size_t)(i0 + m) * D;
  const unsigned short* rAl = Lo + (size_t)(i0 + m) * D;
  const unsigned short* rB0h = Hi + (size_t)(j0 +  0 + m) * D;
  const unsigned short* rB0l = Lo + (size_t)(j0 +  0 + m) * D;
  const unsigned short* rB1h = Hi + (size_t)(j0 + 16 + m) * D;
  const unsigned short* rB1l = Lo + (size_t)(j0 + 16 + m) * D;
  const unsigned short* rB2h = Hi + (size_t)(j0 + 32 + m) * D;
  const unsigned short* rB2l = Lo + (size_t)(j0 + 32 + m) * D;
  const unsigned short* rB3h = Hi + (size_t)(j0 + 48 + m) * D;
  const unsigned short* rB3l = Lo + (size_t)(j0 + 48 + m) * D;

  v8f c0 = {}, c1 = {}, c2 = {}, c3 = {};

  for (int k0 = 0; k0 < D; k0 += 32) {
    // A fragment (16x32 bf16): vgpr v -> K = 8*hh + (v<4 ? 2v : 16+2(v-4))
    BFrag ahi, alo;
    {
      const uint4* pa = (const uint4*)(rAh + k0 + 8 * hh);
      ahi.q[0] = pa[0];
      ahi.q[1] = pa[2];                  // +16 ushort elements = +2 uint4
      const uint4* pl = (const uint4*)(rAl + k0 + 8 * hh);
      alo.q[0] = pl[0];
      alo.q[1] = pl[2];
    }
    // B fragment (32x16 bf16): vgpr v -> K = 8*hh + (v<4 ? 2v : 16+2(v-4))
#define GRAM_STEP(CT, RBH, RBL)                                               \
    do {                                                                      \
      BFrag bhi, blo;                                                         \
      const uint4* pb = (const uint4*)((RBH) + k0 + 8 * hh);                  \
      bhi.q[0] = pb[0];                                                       \
      bhi.q[1] = pb[2];                                                       \
      const uint4* pc = (const uint4*)((RBL) + k0 + 8 * hh);                  \
      blo.q[0] = pc[0];                                                       \
      blo.q[1] = pc[2];                                                       \
      CT = __builtin_amdgcn_wmma_f32_16x16x32_bf16(false, ahi.v, false,       \
              bhi.v, (short)0, CT, false, false);                             \
      CT = __builtin_amdgcn_wmma_f32_16x16x32_bf16(false, ahi.v, false,       \
              blo.v, (short)0, CT, false, false);                             \
      CT = __builtin_amdgcn_wmma_f32_16x16x32_bf16(false, alo.v, false,       \
              bhi.v, (short)0, CT, false, false);                             \
    } while (0)

    GRAM_STEP(c0, rB0h, rB0l);
    GRAM_STEP(c1, rB1h, rB1l);
    GRAM_STEP(c2, rB2h, rB2l);
    GRAM_STEP(c3, rB3h, rB3l);
#undef GRAM_STEP
  }

  // Epilogue: C/D element at vgpr v, lane ln is (M = v + 8*hh, N = ln&15)
  const int n = m;
  float sqj[4], sj[4];
  int lj[4];
#pragma unroll
  for (int t = 0; t < 4; ++t) {
    int j = j0 + 16 * t + n;
    sqj[t] = sq[j];
    sj[t]  = sv[j];
    lj[t]  = labels[j];
  }

  float wsum = 0.f;
#pragma unroll
  for (int v = 0; v < 8; ++v) {
    const int i = i0 + 8 * hh + v;
    const float sqi = sq[i];
    const float si  = sv[i];
    const int   li  = labels[i];
    const float gv[4] = { c0[v], c1[v], c2[v], c3[v] };
#pragma unroll
    for (int t = 0; t < 4; ++t) {
      const int j = j0 + 16 * t + n;
      if (i < j) {
        float d2 = sqi + sqj[t] - 2.f * gv[t] +
                   (2.f * EPS_PD) * (si - sj[t]) + dEps2;
        d2 = fmaxf(d2, 1e-12f);
        float dist = sqrtf(d2);
        float pl;
        if (li == lj[t]) {
          float md = fmaxf(MARGIN - dist, 0.f);
          pl = md * md;
        } else {
          pl = d2;
        }
        wsum += pl;
      }
    }
  }

  for (int off = 16; off > 0; off >>= 1)
    wsum += __shfl_down(wsum, off, 32);
  if (lane == 0) atomicAdd(acc, (double)wsum);
}

// ---------------- kernel 4: cross entropy (one block per row) ----------------
__global__ void ce_kernel(const float* __restrict__ preds,
                          const int* __restrict__ labels,
                          double* __restrict__ acc, int C) {
  __shared__ float red[256];
  const int row = blockIdx.x;
  const float* p = preds + (size_t)row * C;

  float mx = -3.4e38f;
  for (int c = threadIdx.x; c < C; c += blockDim.x) mx = fmaxf(mx, p[c]);
  red[threadIdx.x] = mx;
  __syncthreads();
  for (int off = 128; off > 0; off >>= 1) {
    if ((int)threadIdx.x < off)
      red[threadIdx.x] = fmaxf(red[threadIdx.x], red[threadIdx.x + off]);
    __syncthreads();
  }
  mx = red[0];
  __syncthreads();

  float sum = 0.f;
  for (int c = threadIdx.x; c < C; c += blockDim.x) sum += expf(p[c] - mx);
  red[threadIdx.x] = sum;
  __syncthreads();
  for (int off = 128; off > 0; off >>= 1) {
    if ((int)threadIdx.x < off)
      red[threadIdx.x] += red[threadIdx.x + off];
    __syncthreads();
  }
  if (threadIdx.x == 0) {
    float lse = logf(red[0]) + mx;
    float loss = lse - p[labels[row]];
    atomicAdd(acc + 1, (double)loss);
  }
}

// ---------------- kernel 5: finalize ----------------
__global__ void finalize_kernel(const double* __restrict__ acc,
                                float* __restrict__ out,
                                double inv_npairs, double invN) {
  double contrastive = acc[0] * inv_npairs;
  double supervised  = acc[1] * invN;
  out[0] = (float)(contrastive + supervised);
  out[1] = (float)contrastive;
  out[2] = (float)supervised;
}

extern "C" void kernel_launch(void* const* d_in, const int* in_sizes, int n_in,
                              void* d_out, int out_size, void* d_ws, size_t ws_size,
                              hipStream_t stream) {
  const float* latents = (const float*)d_in[0];
  const int*   labels  = (const int*)d_in[1];
  const float* preds   = (const float*)d_in[2];

  const int N = in_sizes[1];          // 4096
  const int D = in_sizes[0] / N;      // 256
  const int C = in_sizes[2] / N;      // 1000

  // workspace layout
  char* ws = (char*)d_ws;
  double* acc = (double*)ws;                                     // 2 doubles
  float*  sq  = (float*)(ws + 256);                              // N floats
  float*  sv  = sq + N;                                          // N floats
  unsigned short* hi = (unsigned short*)(ws + 256 + (size_t)8 * N);
  unsigned short* lo = hi + (size_t)N * D;

  const float dEps2 = (float)D * EPS_PD * EPS_PD;

  zero_acc_kernel<<<1, 1, 0, stream>>>(acc);

  const int nelem = N * D;
  split_bf16_kernel<<<(nelem + 255) / 256, 256, 0, stream>>>(latents, hi, lo, nelem);

  row_stats_kernel<<<N, 256, 0, stream>>>(latents, sq, sv, D);

  const int jobs = (N / 16) * (N / 64);      // one wave per 16x64 strip
  const int blocks = (jobs + 7) / 8;         // 8 waves (256 threads) per block
  contrastive_kernel<<<blocks, 256, 0, stream>>>(hi, lo, sq, sv, labels, acc,
                                                 N, D, dEps2);

  ce_kernel<<<N, 256, 0, stream>>>(preds, labels, acc, C);

  const double n_pairs = 0.5 * (double)N * (double)(N - 1);
  finalize_kernel<<<1, 1, 0, stream>>>(acc, (float*)d_out,
                                       1.0 / n_pairs, 1.0 / (double)N);
}
